// Colla_Attention_87591563034981
// MI455X (gfx1250) — matrix operand order repaired and checked by reading, not compiled, not run on hardware
//
#include <hip/hip_runtime.h>
#include <hip/hip_bf16.h>

#define BD   2
#define LD   2048
#define DIMD 512
#define HH   8
#define DHD  64
#define NTOK (BD * LD)

typedef __attribute__((ext_vector_type(16))) __bf16 v16bf;
typedef __attribute__((ext_vector_type(8)))  float  v8f;
typedef __attribute__((ext_vector_type(4)))  int    v4i;

#if defined(__gfx1250__) && defined(__has_builtin)
#if __has_builtin(__builtin_amdgcn_global_load_async_to_lds_b128)
#define USE_ASYNC_LDS 1
#endif
#endif

#ifdef USE_ASYNC_LDS
typedef __attribute__((address_space(1))) v4i gv4i;  // global int4
typedef __attribute__((address_space(3))) v4i lv4i;  // LDS int4
static __device__ __forceinline__ void async_cp_b128(const void* src, void* dst) {
    __builtin_amdgcn_global_load_async_to_lds_b128(
        (gv4i*)src, (lv4i*)dst, 0, 0);
}
static __device__ __forceinline__ void wait_async0() {
#if __has_builtin(__builtin_amdgcn_s_wait_asynccnt)
    __builtin_amdgcn_s_wait_asynccnt(0);
#else
    asm volatile("s_wait_asynccnt 0x0" ::: "memory");
#endif
}
#endif

union Frag {
    v16bf v;
    uint4 q[2];
    unsigned d[8];
};

static __device__ __forceinline__ unsigned pkbf2(float a, float b) {
    unsigned ua = __builtin_bit_cast(unsigned, a);
    unsigned ub = __builtin_bit_cast(unsigned, b);
    ua = (ua + 0x7fffu + ((ua >> 16) & 1u)) >> 16;
    ub = (ub + 0x7fffu + ((ub >> 16) & 1u)) >> 16;
    return ua | (ub << 16);
}
static __device__ __forceinline__ unsigned short bf1(float a) {
    unsigned ua = __builtin_bit_cast(unsigned, a);
    return (unsigned short)((ua + 0x7fffu + ((ua >> 16) & 1u)) >> 16);
}

// ---------------------------------------------------------------- converts
__global__ void k_cvt_x(const float* __restrict__ x, unsigned short* __restrict__ xb, int n) {
    int i = blockIdx.x * blockDim.x + threadIdx.x;
    if (i < n) xb[i] = bf1(x[i]);
}

// wt[dout*DIM + din] = bf16(w[dout*DIM + din])
__global__ void k_tr_w(const float* __restrict__ w, unsigned short* __restrict__ wt) {
    int i = blockIdx.x * blockDim.x + threadIdx.x;
    int din = i / DIMD, dout = i % DIMD;
    wt[dout * DIMD + din] = bf1(w[dout * DIMD + din]);
}

// cb[b,h,s] = sum_d x[b,s,d] * Wb[h,d]
__global__ void k_cb(const float* __restrict__ x, const float* __restrict__ Wb,
                     float* __restrict__ cb) {
    int i = blockIdx.x * blockDim.x + threadIdx.x;
    int s = i % LD, bh = i / LD;
    int h = bh % HH, b = bh / HH;
    if (b >= BD) return;
    const float* xr = x + ((size_t)b * LD + s) * DIMD;
    const float* wr = Wb + (size_t)h * DIMD;
    float acc = 0.f;
    for (int d = 0; d < DIMD; ++d) acc += xr[d] * wr[d];
    cb[(size_t)bh * LD + s] = acc;
}

// ---------------------------------------------------------------- projections (WMMA)
// z=0: Q -> qhT[b,h,l,d] = bf16(q * mixing[h,d]);  z=1: K[S,D];  z=2: V[S,D]
__global__ void __launch_bounds__(256) k_proj(const unsigned short* __restrict__ xb,
                                              const unsigned short* __restrict__ WqT,
                                              const unsigned short* __restrict__ WkT,
                                              const unsigned short* __restrict__ WvT,
                                              const float* __restrict__ mixing,
                                              unsigned short* __restrict__ qhT,
                                              unsigned short* __restrict__ Kb,
                                              unsigned short* __restrict__ Vb) {
    const int z    = blockIdx.z;
    const unsigned short* Wt = (z == 0) ? WqT : (z == 1) ? WkT : WvT;
    const int warp = threadIdx.x >> 5;
    const int lane = threadIdx.x & 31;
    const int half = lane >> 4;
    const int m    = lane & 15;
    const int wid  = blockIdx.x * 8 + warp;           // 0..2047
    const int tok0 = (wid & 255) * 16;
    const int nb   = (wid >> 8) * 64;

    v8f acc[4] = {};
#pragma unroll
    for (int kk = 0; kk < 16; ++kk) {
        Frag a;
        const unsigned short* ap = xb + (size_t)(tok0 + m) * DIMD + kk * 32 + half * 8;
        a.q[0] = *(const uint4*)ap;
        a.q[1] = *(const uint4*)(ap + 16);
#pragma unroll
        for (int t = 0; t < 4; ++t) {
            Frag bfr;
            const unsigned short* bp = Wt + (size_t)(nb + t * 16 + m) * DIMD + kk * 32 + half * 8;
            bfr.q[0] = *(const uint4*)bp;
            bfr.q[1] = *(const uint4*)(bp + 16);
            acc[t] = __builtin_amdgcn_wmma_f32_16x16x32_bf16(
                false, a.v, false, bfr.v, (short)0, acc[t], false, false);
        }
    }

    const int b    = tok0 / LD;
    const int lrow = tok0 % LD;
    if (z == 0) {
#pragma unroll
        for (int h = 0; h < HH; ++h) {
#pragma unroll
            for (int t = 0; t < 4; ++t) {
                const int d   = nb + t * 16 + m;
                const float mx = mixing[(size_t)h * DIMD + d];
#pragma unroll
                for (int r = 0; r < 8; ++r)
                    qhT[((size_t)(b * HH + h) * LD + lrow + half * 8 + r) * DIMD + d] = bf1(acc[t][r] * mx);
            }
        }
    } else {
        unsigned short* O = (z == 1) ? Kb : Vb;
#pragma unroll
        for (int t = 0; t < 4; ++t) {
            const int d = nb + t * 16 + m;
#pragma unroll
            for (int r = 0; r < 8; ++r) {
                const int tok = tok0 + r + half * 8;
                O[(size_t)tok * DIMD + d] = bf1(acc[t][r]);
            }
        }
    }
}

// ---------------------------------------------------------------- flash attention (WMMA)
__global__ void __launch_bounds__(256) k_attn(const unsigned short* __restrict__ qhT,
                                              const unsigned short* __restrict__ Kb,
                                              const unsigned short* __restrict__ Vb,
                                              const float* __restrict__ cb,
                                              unsigned short* __restrict__ ctxb) {
    __shared__ __align__(16) unsigned short kTile[2][32 * DIMD];  // 2 x 32KB

    const int warp = threadIdx.x >> 5;
    const int lane = threadIdx.x & 31;
    const int half = lane >> 4;
    const int m    = lane & 15;
    const int bh   = blockIdx.y;
    const int b    = bh >> 3;
    const int h    = bh & 7;
    const int l0   = blockIdx.x * 128 + warp * 16;

    // preload q B-fragments: B[32 d x 16 l] per k-step, from qhT[b,h,l,d]
    Frag qf[16];
    {
        const unsigned short* qbase = qhT + (size_t)bh * DIMD * LD;
#pragma unroll
        for (int kk = 0; kk < 16; ++kk) {
            const unsigned short* p = qbase + (size_t)(l0 + m) * DIMD + kk * 32 + half * 8;
            qf[kk].q[0] = *(const uint4*)p;
            qf[kk].q[1] = *(const uint4*)(p + 16);
        }
    }

    v8f  ctx[4] = {};
    float mrun = -INFINITY, lsum = 0.f;
    const float* cbp = cb + (size_t)bh * LD;
    const unsigned short* kbase = Kb + (size_t)b * LD * DIMD;
    const unsigned short* vbase = Vb + (size_t)b * LD * DIMD + h * DHD;
    const int NIT = LD / 32;  // 64

    // -------- stage tile 0 into LDS (tile = contiguous 32x512 bf16 = 32KB)
#ifdef USE_ASYNC_LDS
    {
        const unsigned short* src = kbase + threadIdx.x * 64;
        unsigned short*       dst = &kTile[0][0] + threadIdx.x * 64;
#pragma unroll
        for (int i = 0; i < 8; ++i) async_cp_b128(src + i * 8, dst + i * 8);
        wait_async0();
    }
    __syncthreads();
#else
    uint4 stg[8];
    {
        const uint4* src = (const uint4*)kbase;
#pragma unroll
        for (int i = 0; i < 8; ++i) stg[i] = src[threadIdx.x * 8 + i];
        uint4* dst = (uint4*)kTile[0];
#pragma unroll
        for (int i = 0; i < 8; ++i) dst[threadIdx.x * 8 + i] = stg[i];
    }
    __syncthreads();
#endif

    const float C = 1.44269504088896340736f;  // log2(e)
    for (int it = 0; it < NIT; ++it) {
        const int s0 = it * 32;
#ifdef USE_ASYNC_LDS
        if (it + 1 < NIT) {  // async DMA next tile into the other buffer
            const unsigned short* src = kbase + (size_t)(s0 + 32) * DIMD + threadIdx.x * 64;
            unsigned short*       dst = &kTile[(it + 1) & 1][0] + threadIdx.x * 64;
#pragma unroll
            for (int i = 0; i < 8; ++i) async_cp_b128(src + i * 8, dst + i * 8);
        }
#else
        if (it + 1 < NIT) {  // prefetch next tile into registers
            const uint4* src = (const uint4*)(kbase + (size_t)(s0 + 32) * DIMD);
#pragma unroll
            for (int i = 0; i < 8; ++i) stg[i] = src[threadIdx.x * 8 + i];
        }
#endif

        // scoresT = K[s-tile, :] x qh^T  -> two 16x16 tiles (s0..15, s16..31)
        const unsigned short* kl = kTile[it & 1];
        v8f accA = {}, accB = {};
#pragma unroll
        for (int kk = 0; kk < 16; ++kk) {
            Frag ka, kc;
            const unsigned short* p0 = kl + (size_t)m * DIMD + kk * 32 + half * 8;
            ka.q[0] = *(const uint4*)p0;
            ka.q[1] = *(const uint4*)(p0 + 16);
            const unsigned short* p1 = p0 + 16 * DIMD;
            kc.q[0] = *(const uint4*)p1;
            kc.q[1] = *(const uint4*)(p1 + 16);
            accA = __builtin_amdgcn_wmma_f32_16x16x32_bf16(
                false, ka.v, false, qf[kk].v, (short)0, accA, false, false);
            accB = __builtin_amdgcn_wmma_f32_16x16x32_bf16(
                false, kc.v, false, qf[kk].v, (short)0, accB, false, false);
        }

        // content bias + 1/sqrt(dh) scale
        float4 c0  = *(const float4*)(cbp + s0 + half * 8);
        float4 c0b = *(const float4*)(cbp + s0 + half * 8 + 4);
        float4 c1  = *(const float4*)(cbp + s0 + 16 + half * 8);
        float4 c1b = *(const float4*)(cbp + s0 + 16 + half * 8 + 4);
        float cbA[8] = {c0.x, c0.y, c0.z, c0.w, c0b.x, c0b.y, c0b.z, c0b.w};
        float cbB[8] = {c1.x, c1.y, c1.z, c1.w, c1b.x, c1b.y, c1b.z, c1b.w};

        float pA[8], pB[8], mloc = -INFINITY;
#pragma unroll
        for (int r = 0; r < 8; ++r) {
            pA[r] = (accA[r] + cbA[r]) * 0.125f;
            pB[r] = (accB[r] + cbB[r]) * 0.125f;
            mloc  = fmaxf(mloc, fmaxf(pA[r], pB[r]));
        }
        mloc = fmaxf(mloc, __shfl_xor(mloc, 16, 32));
        const float mnew = fmaxf(mrun, mloc);
        const float corr = exp2f((mrun - mnew) * C);
        float tsum = 0.f;
#pragma unroll
        for (int r = 0; r < 8; ++r) {
            pA[r] = exp2f((pA[r] - mnew) * C);
            pB[r] = exp2f((pB[r] - mnew) * C);
            tsum += pA[r] + pB[r];
        }
        tsum += __shfl_xor(tsum, 16, 32);
        lsum = lsum * corr + tsum;
        mrun = mnew;

        // rescale ctx accumulators (per query-row factor, fetched via bpermute)
#pragma unroll
        for (int r = 0; r < 8; ++r) {
            const float f = __shfl(corr, r + half * 8, 32);
            ctx[0][r] *= f; ctx[1][r] *= f; ctx[2][r] *= f; ctx[3][r] *= f;
        }

        // transposed-probs C-layout == probs A-fragment layout: just pack to bf16
        Frag pa;
#pragma unroll
        for (int j = 0; j < 4; ++j) {
            pa.d[j]     = pkbf2(pA[2 * j], pA[2 * j + 1]);
            pa.d[4 + j] = pkbf2(pB[2 * j], pB[2 * j + 1]);
        }

        // ctx += probs x V
#pragma unroll
        for (int t = 0; t < 4; ++t) {
            Frag vb;
            const unsigned short* vp = vbase + (size_t)(s0 + half * 8) * DIMD + t * 16 + m;
#pragma unroll
            for (int j = 0; j < 4; ++j) {
                vb.d[j]     = (unsigned)vp[(2 * j) * DIMD] | ((unsigned)vp[(2 * j + 1) * DIMD] << 16);
                vb.d[4 + j] = (unsigned)vp[(16 + 2 * j) * DIMD] | ((unsigned)vp[(17 + 2 * j) * DIMD] << 16);
            }
            ctx[t] = __builtin_amdgcn_wmma_f32_16x16x32_bf16(
                false, pa.v, false, vb.v, (short)0, ctx[t], false, false);
        }

#ifdef USE_ASYNC_LDS
        if (it + 1 < NIT) wait_async0();
        __syncthreads();
#else
        __syncthreads();
        if (it + 1 < NIT) {
            uint4* dst = (uint4*)kTile[(it + 1) & 1];
#pragma unroll
            for (int i = 0; i < 8; ++i) dst[threadIdx.x * 8 + i] = stg[i];
            __syncthreads();
        }
#endif
    }

    // normalize and store ctx (bf16 [B,L,D])
    const float rinv = 1.f / lsum;
#pragma unroll
    for (int r = 0; r < 8; ++r) {
        const float f  = __shfl(rinv, r + half * 8, 32);
        const int   lg = l0 + r + half * 8;
#pragma unroll
        for (int t = 0; t < 4; ++t) {
            ctxb[((size_t)b * LD + lg) * DIMD + h * DHD + t * 16 + m] = bf1(ctx[t][r] * f);
        }
    }
}

// ---------------------------------------------------------------- output projection (WMMA)
__global__ void __launch_bounds__(256) k_oproj(const unsigned short* __restrict__ ctxb,
                                               const unsigned short* __restrict__ WdT,
                                               const float* __restrict__ bd,
                                               float* __restrict__ out) {
    const int warp = threadIdx.x >> 5;
    const int lane = threadIdx.x & 31;
    const int half = lane >> 4;
    const int m    = lane & 15;
    const int wid  = blockIdx.x * 8 + warp;
    const int tok0 = (wid & 255) * 16;
    const int nb   = (wid >> 8) * 64;

    v8f acc[4] = {};
#pragma unroll
    for (int kk = 0; kk < 16; ++kk) {
        Frag a;
        const unsigned short* ap = ctxb + (size_t)(tok0 + m) * DIMD + kk * 32 + half * 8;
        a.q[0] = *(const uint4*)ap;
        a.q[1] = *(const uint4*)(ap + 16);
#pragma unroll
        for (int t = 0; t < 4; ++t) {
            Frag bfr;
            const unsigned short* bp = WdT + (size_t)(nb + t * 16 + m) * DIMD + kk * 32 + half * 8;
            bfr.q[0] = *(const uint4*)bp;
            bfr.q[1] = *(const uint4*)(bp + 16);
            acc[t] = __builtin_amdgcn_wmma_f32_16x16x32_bf16(
                false, a.v, false, bfr.v, (short)0, acc[t], false, false);
        }
    }
#pragma unroll
    for (int t = 0; t < 4; ++t) {
        const int d    = nb + t * 16 + m;
        const float bi = bd[d];
#pragma unroll
        for (int r = 0; r < 8; ++r)
            out[(size_t)(tok0 + r + half * 8) * DIMD + d] = acc[t][r] + bi;
    }
}

// ---------------------------------------------------------------- host launcher
extern "C" void kernel_launch(void* const* d_in, const int* in_sizes, int n_in,
                              void* d_out, int out_size, void* d_ws, size_t ws_size,
                              hipStream_t stream) {
    (void)in_sizes; (void)n_in; (void)out_size; (void)ws_size;
    const float* x      = (const float*)d_in[0];
    const float* Wq     = (const float*)d_in[1];
    const float* Wk     = (const float*)d_in[2];
    const float* Wv     = (const float*)d_in[3];
    const float* Wb     = (const float*)d_in[4];
    const float* mixing = (const float*)d_in[5];
    const float* Wd     = (const float*)d_in[6];
    const float* bd     = (const float*)d_in[7];
    float* out = (float*)d_out;

    char* ws   = (char*)d_ws;
    size_t off = 0;
    auto take = [&](size_t bytes) {
        char* p = ws + off;
        off = (off + bytes + 255) & ~(size_t)255;
        return p;
    };
    unsigned short* XB   = (unsigned short*)take((size_t)NTOK * DIMD * 2);
    unsigned short* WQT  = (unsigned short*)take((size_t)DIMD * DIMD * 2);
    unsigned short* WKT  = (unsigned short*)take((size_t)DIMD * DIMD * 2);
    unsigned short* WVT  = (unsigned short*)take((size_t)DIMD * DIMD * 2);
    unsigned short* WDT  = (unsigned short*)take((size_t)DIMD * DIMD * 2);
    unsigned short* QHT  = (unsigned short*)take((size_t)BD * HH * DIMD * LD * 2);
    unsigned short* KB   = (unsigned short*)take((size_t)NTOK * DIMD * 2);
    unsigned short* VB   = (unsigned short*)take((size_t)NTOK * DIMD * 2);
    unsigned short* CTXB = (unsigned short*)take((size_t)NTOK * DIMD * 2);
    float*          CB   = (float*)take((size_t)BD * HH * LD * 4);

    k_cvt_x<<<dim3((NTOK * DIMD) / 256), 256, 0, stream>>>(x, XB, NTOK * DIMD);
    k_tr_w<<<dim3((DIMD * DIMD) / 256), 256, 0, stream>>>(Wq, WQT);
    k_tr_w<<<dim3((DIMD * DIMD) / 256), 256, 0, stream>>>(Wk, WKT);
    k_tr_w<<<dim3((DIMD * DIMD) / 256), 256, 0, stream>>>(Wv, WVT);
    k_tr_w<<<dim3((DIMD * DIMD) / 256), 256, 0, stream>>>(Wd, WDT);
    k_cb<<<dim3((BD * HH * LD) / 256), 256, 0, stream>>>(x, Wb, CB);
    k_proj<<<dim3(256, 1, 3), 256, 0, stream>>>(XB, WQT, WKT, WVT, mixing, QHT, KB, VB);
    k_attn<<<dim3(LD / 128, BD * HH), 256, 0, stream>>>(QHT, KB, VB, CB, CTXB);
    k_oproj<<<dim3(256), 256, 0, stream>>>(CTXB, WDT, bd, out);
}
